// GraphFlowModel_38165079392412
// MI455X (gfx1250) — matrix operand order repaired and checked by reading, not compiled, not run on hardware
//
#include <hip/hip_runtime.h>

// GraphFlowModel on MI455X (gfx1250): per-node gather + 2-layer MLP + projection.
// Transposed GEMMs (h1^T = W1^T x X^T, h2^T = W2^T x h1^T): C-layouts map to
// contiguous LDS b128 traffic, biases init the WMMA accumulators directly,
// final GEMV reduces with one shfl_xor. bf16 WMMA, f32 accumulate.
// All WMMA operands are preloaded so the WMMA bursts issue without WAR nops.

typedef __attribute__((ext_vector_type(16))) __bf16 v16bf;
typedef __attribute__((ext_vector_type(8)))  float  v8f;

#define NEG_SLOPE 0.01f

constexpr int BATCH  = 512;
constexpr int DDIM   = 32;   // K of GEMM1
constexpr int HDIM   = 64;   // M of GEMM1 / M,K of GEMM2
constexpr int CHUNK  = 128;  // batch rows per chunk
constexpr int NCHUNK = BATCH / CHUNK;   // 4

__device__ __forceinline__ float leaky(float x) {
    return __builtin_amdgcn_fmed3f(x, NEG_SLOPE * x, __builtin_inff());
}

union BFrag { uint4  u[2]; v16bf v; };   // 16 bf16 = one WMMA A/B operand
union F8    { float4 f[2]; v8f   v; };   // 8 f32  = one WMMA C/D operand
union H8    { __bf16 h[8]; uint4 u; };   // 8 bf16 packed for one ds_store_b128

// ---- pre-pass: state f32 -> bf16 (no per-gather v_cvt, half the gather bytes)
__global__ __launch_bounds__(256) void gfm_state_bf16_kernel(
    const float* __restrict__ src, __bf16* __restrict__ dst, int total8)
{
    int i = blockIdx.x * 256 + threadIdx.x;
    if (i < total8) {
        const float4 a = ((const float4*)src)[2 * i];
        const float4 b = ((const float4*)src)[2 * i + 1];
        H8 o;
        o.h[0] = (__bf16)a.x; o.h[1] = (__bf16)a.y;
        o.h[2] = (__bf16)a.z; o.h[3] = (__bf16)a.w;
        o.h[4] = (__bf16)b.x; o.h[5] = (__bf16)b.y;
        o.h[6] = (__bf16)b.z; o.h[7] = (__bf16)b.w;
        ((uint4*)dst)[i] = o.u;
    }
}

template <bool USE_BF>
__global__ __launch_bounds__(256) void gfm_node_kernel(
    const float* __restrict__ state,       // (B, NN) f32 (USE_BF==false)
    const __bf16* __restrict__ stateBF,    // (B, NN) bf16 (USE_BF==true)
    const long long* __restrict__ adj,     // (NN, D) int64
    const float* __restrict__ W1,          // (NN, D, H)
    const float* __restrict__ b1,          // (NN, H)
    const float* __restrict__ W2,          // (NN, H, H)  [h = K, k = out col]
    const float* __restrict__ b2,          // (NN, H)
    const float* __restrict__ W3,          // (NN, H)
    float* __restrict__ outp,              // staging (NN,B) or direct (B,NN)
    int NN, long long sb, long long sn)    // element strides: addr = b*sb + n*sn
{
    const int n    = blockIdx.x;
    const int t    = threadIdx.x;
    const int lane = t & 31;
    const int w    = t >> 5;          // 8 waves
    const int c    = lane & 15;
    const int kh   = lane >> 4;       // lane half

    __shared__ __align__(16) __bf16 sW1T[HDIM * DDIM];   // [h][d]  4 KB
    __shared__ __align__(16) __bf16 sW2T[HDIM * HDIM];   // [k][h]  8 KB
    __shared__ __align__(16) __bf16 sX  [CHUNK * DDIM];  // [row][d] 8 KB
    __shared__ __align__(16) __bf16 sH1 [CHUNK * HDIM];  // [row][h] 16 KB (wave-private rows)
    __shared__ __align__(16) float  sB1[HDIM];
    __shared__ __align__(16) float  sB2[HDIM];
    __shared__ __align__(16) float  sW3[HDIM];
    __shared__ int sAdj[DDIM];

    // ---- stage node-local weights (bf16, transposed) ----
    if (t < DDIM) sAdj[t] = (int)adj[(long long)n * DDIM + t];
    if (t < HDIM) {
        sB1[t] = b1[(long long)n * HDIM + t];
        sB2[t] = b2[(long long)n * HDIM + t];
        sW3[t] = W3[(long long)n * HDIM + t];
    }
    {
        const float* w1p = W1 + (long long)n * (DDIM * HDIM);
        for (int e = t; e < DDIM * HDIM; e += 256) {      // W1[n][d][h] -> W1T[h][d]
            int d = e >> 6, h = e & 63;
            sW1T[h * DDIM + d] = (__bf16)w1p[e];
        }
        const float* w2p = W2 + (long long)n * (HDIM * HDIM);
        for (int e = t; e < HDIM * HDIM; e += 256) {      // W2[n][h][k] -> W2T[k][h]
            int h = e >> 6, k = e & 63;
            sW2T[k * HDIM + h] = (__bf16)w2p[e];
        }
    }
    __syncthreads();

    // gather assignment: thread owns column d, rows r0, r0+8, ... (all offsets
    // become compile-time immediates off one base address; 32-bit arithmetic)
    const int gd  = t & 31;
    const int gr0 = t >> 5;                  // 0..7
    const int gcol = sAdj[gd];
    const int gbase = gr0 * NN + gcol;       // element offset (fits in 31 bits)

    for (int ch = 0; ch < NCHUNK; ++ch) {
        const int rbase = ch * CHUNK;

        // ---- gather chunk: X[r][d] = state[rbase+r][adj[n][d]] ----
        {
            const int src0 = gbase + rbase * NN;
            const int lds0 = gr0 * DDIM + gd;
            #pragma unroll
            for (int rr = 0; rr < CHUNK / 8; ++rr) {
                if (USE_BF) sX[lds0 + rr * 8 * DDIM] = stateBF[src0 + rr * 8 * NN];
                else        sX[lds0 + rr * 8 * DDIM] = (__bf16)state[src0 + rr * 8 * NN];
            }
        }
        __syncthreads();

        const int m0 = w * 16;   // this wave's 16 batch rows within the chunk

        // ==== GEMM1 (transposed): h1^T = W1^T (A) x X^T (B) + b1 ====
        BFrag xb;   // B operand: lane c = batch row m0+c, K = d
        {
            const char* p = (const char*)sX + (m0 + c) * (DDIM * 2) + kh * 16;
            xb.u[0] = *(const uint4*)(p);
            xb.u[1] = *(const uint4*)(p + 32);
        }
        BFrag a1[4];
        F8    c1[4];
        #pragma unroll
        for (int mt = 0; mt < 4; ++mt) {     // preload ALL operands first
            const char* ap = (const char*)sW1T + (mt * 16 + c) * (DDIM * 2);
            a1[mt].u[0] = *(const uint4*)(ap + kh * 16);
            a1[mt].u[1] = *(const uint4*)(ap + 32 + kh * 16);
            const float* bp = sB1 + mt * 16 + 8 * kh;    // C init = b1[mt*16+8*kh+v]
            c1[mt].f[0] = *(const float4*)(bp);
            c1[mt].f[1] = *(const float4*)(bp + 4);
        }
        v8f d1[4];
        #pragma unroll
        for (int mt = 0; mt < 4; ++mt)       // back-to-back burst, no WAR on sources
            d1[mt] = __builtin_amdgcn_wmma_f32_16x16x32_bf16(
                false, a1[mt].v, false, xb.v, (short)0, c1[mt].v, false, false);
        #pragma unroll
        for (int mt = 0; mt < 4; ++mt) {     // epilogue: leaky -> packed bf16 -> b128
            // C layout: lane c = batch row m0+c, VGPR v -> h = mt*16 + 8*kh + v
            H8 o;
            #pragma unroll
            for (int v = 0; v < 8; ++v) o.h[v] = (__bf16)leaky(d1[mt][v]);
            *(uint4*)((char*)sH1 + (m0 + c) * (HDIM * 2) + mt * 32 + kh * 16) = o.u;
        }
        // no barrier: sH1 rows m0..m0+15 produced AND consumed by this wave only

        // ==== GEMM2 (transposed): h2^T = W2^T (A) x h1^T (B) + b2; fused GEMV3 ====
        BFrag hb[2];   // B operand: lane c = batch row m0+c, K = h
        #pragma unroll
        for (int kc = 0; kc < 2; ++kc) {
            const char* p = (const char*)sH1 + (m0 + c) * (HDIM * 2) + kc * 64 + kh * 16;
            hb[kc].u[0] = *(const uint4*)(p);
            hb[kc].u[1] = *(const uint4*)(p + 32);
        }
        BFrag a2[8];
        v8f   d2[4];
        #pragma unroll
        for (int mt = 0; mt < 4; ++mt) {     // preload all 8 A-frags + bias inits
            const char* ap = (const char*)sW2T + (mt * 16 + c) * (HDIM * 2);
            #pragma unroll
            for (int kc = 0; kc < 2; ++kc) {
                a2[mt * 2 + kc].u[0] = *(const uint4*)(ap + kc * 64 + kh * 16);
                a2[mt * 2 + kc].u[1] = *(const uint4*)(ap + kc * 64 + 32 + kh * 16);
            }
            F8 acc;   // C init = b2[k], k = mt*16 + 8*kh + v
            const float* bp = sB2 + mt * 16 + 8 * kh;
            acc.f[0] = *(const float4*)(bp);
            acc.f[1] = *(const float4*)(bp + 4);
            d2[mt] = acc.v;
        }
        #pragma unroll
        for (int mt = 0; mt < 4; ++mt) {     // 8 WMMAs, 4 independent accum chains
            d2[mt] = __builtin_amdgcn_wmma_f32_16x16x32_bf16(
                false, a2[mt * 2 + 0].v, false, hb[0].v, (short)0, d2[mt], false, false);
            d2[mt] = __builtin_amdgcn_wmma_f32_16x16x32_bf16(
                false, a2[mt * 2 + 1].v, false, hb[1].v, (short)0, d2[mt], false, false);
        }
        float racc = 0.0f;
        #pragma unroll
        for (int mt = 0; mt < 4; ++mt) {
            F8 w3v;   // W3[k], per-VGPR uniform
            const float* wp = sW3 + mt * 16 + 8 * kh;
            w3v.f[0] = *(const float4*)(wp);
            w3v.f[1] = *(const float4*)(wp + 4);
            #pragma unroll
            for (int v = 0; v < 8; ++v)
                racc = fmaf(leaky(d2[mt][v]), w3v.v[v], racc);   // dot over k (this half)
        }
        // lane L and L^16 hold the two k-halves of the same batch row: one shuffle
        racc += __shfl_xor(racc, 16, 32);
        if (lane < 16) {   // 16 coalesced per-lane stores
            long long b = rbase + m0 + c;
            outp[b * sb + (long long)n * sn] = leaky(racc);
        }
        __syncthreads();   // protect sX before next chunk's gather
    }
}

// (NN,B) -> (B,NN) tiled transpose for coalesced final output
__global__ __launch_bounds__(256) void gfm_transpose_kernel(
    const float* __restrict__ src, float* __restrict__ dst, int NN)
{
    __shared__ float tile[32][33];
    const int tx = threadIdx.x, ty = threadIdx.y;  // 32 x 8
    const int b = blockIdx.y * 32 + tx;
    #pragma unroll
    for (int j = 0; j < 4; ++j) {
        int nn = blockIdx.x * 32 + ty + j * 8;
        if (nn < NN) tile[ty + j * 8][tx] = src[(long long)nn * BATCH + b];
    }
    __syncthreads();
    const int nn2 = blockIdx.x * 32 + tx;
    #pragma unroll
    for (int j = 0; j < 4; ++j) {
        int b2 = blockIdx.y * 32 + ty + j * 8;
        if (nn2 < NN) dst[(long long)b2 * NN + nn2] = tile[tx][ty + j * 8];
    }
}

extern "C" void kernel_launch(void* const* d_in, const int* in_sizes, int n_in,
                              void* d_out, int out_size, void* d_ws, size_t ws_size,
                              hipStream_t stream) {
    const float*     state = (const float*)d_in[0];
    const long long* adj   = (const long long*)d_in[1];
    const float*     W1    = (const float*)d_in[2];
    const float*     b1    = (const float*)d_in[3];
    const float*     W2    = (const float*)d_in[4];
    const float*     b2    = (const float*)d_in[5];
    const float*     W3    = (const float*)d_in[6];
    float* out = (float*)d_out;

    const int NN = in_sizes[1] / DDIM;   // adj is (NN, 32)
    const size_t bfBytes    = (size_t)BATCH * NN * sizeof(__bf16);
    const size_t bfPad      = (bfBytes + 255) & ~(size_t)255;
    const size_t stageBytes = (size_t)NN * BATCH * sizeof(float);

    char* ws = (char*)d_ws;
    const __bf16* stateBF = nullptr;
    float*        stage   = nullptr;

    if (ws_size >= bfPad + stageBytes) {
        __bf16* sbf = (__bf16*)ws;
        stage = (float*)(ws + bfPad);
        const int total8 = (BATCH * NN) / 8;   // BATCH=512 -> divisible
        gfm_state_bf16_kernel<<<(total8 + 255) / 256, 256, 0, stream>>>(state, sbf, total8);
        stateBF = sbf;
    } else if (ws_size >= stageBytes) {
        stage = (float*)ws;
    }

    if (stage) {
        if (stateBF)
            gfm_node_kernel<true><<<NN, 256, 0, stream>>>(state, stateBF, adj, W1, b1, W2, b2, W3,
                                                          stage, NN, 1LL, (long long)BATCH);
        else
            gfm_node_kernel<false><<<NN, 256, 0, stream>>>(state, nullptr, adj, W1, b1, W2, b2, W3,
                                                           stage, NN, 1LL, (long long)BATCH);
        dim3 grid((NN + 31) / 32, BATCH / 32), blk(32, 8);
        gfm_transpose_kernel<<<grid, blk, 0, stream>>>(stage, out, NN);
    } else {
        // fallback: direct (scattered) stores into (B, NN) output
        gfm_node_kernel<false><<<NN, 256, 0, stream>>>(state, nullptr, adj, W1, b1, W2, b2, W3,
                                                       out, NN, (long long)NN, 1LL);
    }
}
